// PolicyNetwork_17549236371854
// MI455X (gfx1250) — matrix operand order repaired and checked by reading, not compiled, not run on hardware
//
#include <hip/hip_runtime.h>
#include <math.h>

#define NNODES 262144
#define GRIDW  512

typedef _Float16 h16;
typedef _Float16 v16h __attribute__((ext_vector_type(16)));
typedef _Float16 v8h  __attribute__((ext_vector_type(8)));
typedef float    v8f  __attribute__((ext_vector_type(8)));

// ---------------------------------------------------------------------------
// Pack f32 weight(s) (K x 128 row-major) into f16 W^T (128 x Ktot row-major),
// optionally concatenating two matrices along K with zero padding to x32.
// ---------------------------------------------------------------------------
__global__ void pack_wt_kernel(const float* __restrict__ Wl, int Kl, int KlPad,
                               const float* __restrict__ Wr, int Kr, int KrPad,
                               h16* __restrict__ out)
{
    int Ktot = KlPad + KrPad;
    int idx  = blockIdx.x * blockDim.x + threadIdx.x;
    if (idx >= 128 * Ktot) return;
    int n = idx / Ktot;
    int k = idx - n * Ktot;
    float v = 0.f;
    if (k < KlPad) {
        if (k < Kl) v = Wl[k * 128 + n];
    } else {
        int kr = k - KlPad;
        if (kr < Kr) v = Wr[kr * 128 + n];
    }
    out[n * Ktot + k] = (h16)v;
}

// ---------------------------------------------------------------------------
// Convert x (N x 26 f32) into cols [32..63] of bufA1 (N x 64 f16), zero pad.
// ---------------------------------------------------------------------------
__global__ void convert_x_kernel(const float* __restrict__ x, h16* __restrict__ dst)
{
    int idx  = blockIdx.x * blockDim.x + threadIdx.x;   // N*32
    int col  = idx & 31;
    int node = idx >> 5;
    if (node >= NNODES) return;
    float v = (col < 26) ? x[node * 26 + col] : 0.f;
    dst[(size_t)node * 64 + 32 + col] = (h16)v;
}

// ---------------------------------------------------------------------------
// Sinusoidal 2D positional encoding into bufPE (N x 32 f16): dims 0..7 rows,
// 8..15 cols, 16..31 zero pad.  div[k] = exp(-k*ln(10000)/4), k=0..3.
// ---------------------------------------------------------------------------
__global__ void pe_kernel(h16* __restrict__ dst)
{
    int idx  = blockIdx.x * blockDim.x + threadIdx.x;   // N*32
    int col  = idx & 31;
    int node = idx >> 5;
    if (node >= NNODES) return;
    float v = 0.f;
    if (col < 16) {
        int axis = col >> 3;                 // 0 = row encoding, 1 = col encoding
        int k    = (col & 7) >> 1;           // 0..3
        int sc   = col & 1;                  // 0 = sin, 1 = cos
        float p  = axis ? (float)(node & (GRIDW - 1)) : (float)(node >> 9);
        float dv = expf(-(float)k * (logf(10000.f) * 0.25f));
        float a  = p * dv;
        v = sc ? cosf(a) : sinf(a);
    }
    dst[(size_t)node * 32 + col] = (h16)v;
}

// ---------------------------------------------------------------------------
// 8-neighbor Moore-grid mean aggregation (SAGE mean):  agg[i] = mean_j h[j].
// One thread per (node, 8-channel chunk). deg = 3/5/8 by position.
// ---------------------------------------------------------------------------
__global__ void stencil_kernel(const h16* __restrict__ src, int srcStride,
                               h16* __restrict__ dst, int dstStride, int C)
{
    int chunks = C >> 3;
    int gid = blockIdx.x * blockDim.x + threadIdx.x;
    if (gid >= NNODES * chunks) return;
    int chunk = gid % chunks;
    int node  = gid / chunks;
    int r = node >> 9;
    int c = node & (GRIDW - 1);

    float acc[8];
#pragma unroll
    for (int j = 0; j < 8; ++j) acc[j] = 0.f;
    int cnt = 0;
#pragma unroll
    for (int dr = -1; dr <= 1; ++dr) {
#pragma unroll
        for (int dc = -1; dc <= 1; ++dc) {
            if (dr == 0 && dc == 0) continue;
            int nr = r + dr, nc = c + dc;
            if ((unsigned)nr < (unsigned)GRIDW && (unsigned)nc < (unsigned)GRIDW) {
                ++cnt;
                const h16* p = src + (size_t)(nr * GRIDW + nc) * srcStride + chunk * 8;
                v8h v = *(const v8h*)p;
#pragma unroll
                for (int j = 0; j < 8; ++j) acc[j] += (float)v[j];
            }
        }
    }
    float inv = 1.f / (float)cnt;
    v8h o;
#pragma unroll
    for (int j = 0; j < 8; ++j) o[j] = (h16)(acc[j] * inv);
    *(v8h*)(dst + (size_t)node * dstStride + chunk * 8) = o;
}

// ---------------------------------------------------------------------------
// WMMA GEMM:  out(M x 128, f16) = relu?( [A0 | A1](M x (K0+K1), f16) @ W^T + b )
//   Wt: 128 x Ktot f16 row-major (output channel major).
//   The ENTIRE W^T (<= 64KB) is staged into dynamic LDS once at kernel entry,
//   so the K loop has no barriers at all.  8 waves/block; each wave owns a
//   16-row M tile x 128 cols (8 WMMA n-tiles).  A fragments are loaded from
//   global with the ISA-documented 16-bit A layout (2 x b128 per lane / step);
//   B fragments are software-pipelined ds_load_b128 pairs (tile t+1 preloaded
//   while WMMA t issues).
// ---------------------------------------------------------------------------
__global__ __launch_bounds__(256) void gemm_wmma_kernel(
    const h16* __restrict__ A0, int K0,
    const h16* __restrict__ A1, int K1,
    const h16* __restrict__ Wt,
    const float* __restrict__ bias,
    h16* __restrict__ out, int outStride,
    int relu)
{
    const int Ktot = K0 + K1;
    extern __shared__ h16 lw[];                // 128 * Ktot halves (<= 64KB)

    const int tid  = threadIdx.x;
    const int wave = tid >> 5;
    const int lane = tid & 31;
    const int g    = lane >> 4;                // lane group 0/1
    const int lr   = lane & 15;                // row-in-tile / col-in-tile
    const int rowBase = blockIdx.x * 128 + wave * 16;

    // ---- stage all of W^T into LDS once (flat copy, v8h chunks)
    {
        const int totalHalves = 128 * Ktot;
        for (int i = tid * 8; i < totalHalves; i += 256 * 8)
            *(v8h*)(lw + i) = *(const v8h*)(Wt + i);
    }
    __syncthreads();

    v8f acc[8];
#pragma unroll
    for (int t = 0; t < 8; ++t)
#pragma unroll
        for (int r = 0; r < 8; ++r) acc[t][r] = 0.f;

    for (int k0 = 0; k0 < Ktot; k0 += 32) {
        // ---- A fragment (16x32 f16): lane holds row lr; K per ISA A-layout
        const h16* Abase;
        int kk;
        if (k0 < K0) { Abase = A0 + (size_t)(rowBase + lr) * K0; kk = k0; }
        else         { Abase = A1 + (size_t)(rowBase + lr) * K1; kk = k0 - K0; }
        v8h alo = *(const v8h*)(Abase + kk + 8 * g);        // K = 8g .. 8g+7
        v8h ahi = *(const v8h*)(Abase + kk + 16 + 8 * g);   // K = 16+8g .. 23+8g
        v16h a;
#pragma unroll
        for (int j = 0; j < 8; ++j) { a[j] = alo[j]; a[j + 8] = ahi[j]; }

        // ---- B fragments (32x16): lane holds col lr, K = 8g + 0..7, 16 + 8g + 0..7
        // software pipeline: preload tile t+1 while WMMA t issues
        const h16* lbase = lw + k0 + 8 * g;
        v8h blo = *(const v8h*)(lbase + (size_t)lr * Ktot);
        v8h bhi = *(const v8h*)(lbase + (size_t)lr * Ktot + 16);
#pragma unroll
        for (int t = 0; t < 8; ++t) {
            v16h b;
#pragma unroll
            for (int j = 0; j < 8; ++j) { b[j] = blo[j]; b[j + 8] = bhi[j]; }
            if (t < 7) {
                int n = (t + 1) * 16 + lr;
                blo = *(const v8h*)(lbase + (size_t)n * Ktot);
                bhi = *(const v8h*)(lbase + (size_t)n * Ktot + 16);
            }
            acc[t] = __builtin_amdgcn_wmma_f32_16x16x32_f16(
                false, a, false, b, (short)0, acc[t], false, false);
        }
    }

    // epilogue: bias + optional relu, store f16.  C layout: M = r + 8g, N = lr
#pragma unroll
    for (int t = 0; t < 8; ++t) {
        int n = t * 16 + lr;
        float bv = bias[n];
#pragma unroll
        for (int r = 0; r < 8; ++r) {
            int row = rowBase + r + 8 * g;
            float v = acc[t][r] + bv;
            if (relu) v = fmaxf(v, 0.f);
            out[(size_t)row * outStride + n] = (h16)v;
        }
    }
}

// ---------------------------------------------------------------------------
// Global mean of node_emb = [h3(128) | pe(16)] -> g[144] (f32)
// ---------------------------------------------------------------------------
__global__ void mean_kernel(const h16* __restrict__ h3, const h16* __restrict__ pe,
                            float* __restrict__ g)
{
    __shared__ float sm[256];
    int dim = blockIdx.x;                     // 0..143
    const h16* base;
    int stride, col;
    if (dim < 128) { base = h3; stride = 128; col = dim; }
    else           { base = pe; stride = 32;  col = dim - 128; }
    float s = 0.f;
    for (int i = threadIdx.x; i < NNODES; i += 256)
        s += (float)base[(size_t)i * stride + col];
    sm[threadIdx.x] = s;
    __syncthreads();
    for (int o = 128; o > 0; o >>= 1) {
        if (threadIdx.x < o) sm[threadIdx.x] += sm[threadIdx.x + o];
        __syncthreads();
    }
    if (threadIdx.x == 0) g[dim] = sm[0] / (float)NNODES;
}

// ---------------------------------------------------------------------------
// Tiny heads on the pooled vector g: value (cr) and action-type logits (at).
// One block, 128 threads, scalar math (negligible work).
// ---------------------------------------------------------------------------
__global__ void heads_kernel(const float* __restrict__ g,
                             const float* atW0, const float* atb0,
                             const float* atW1, const float* atb1,
                             const float* atW2, const float* atb2,
                             const float* crW0, const float* crb0,
                             const float* crW1, const float* crb1,
                             const float* crW2, const float* crb2,
                             float* __restrict__ out)
{
    __shared__ float h0[128];
    __shared__ float h1[128];
    int n = threadIdx.x;

    // ---- value head (cr)
    float a = crb0[n];
    for (int k = 0; k < 144; ++k) a += g[k] * crW0[k * 128 + n];
    h0[n] = fmaxf(a, 0.f);
    __syncthreads();
    a = crb1[n];
    for (int k = 0; k < 128; ++k) a += h0[k] * crW1[k * 128 + n];
    h1[n] = fmaxf(a, 0.f);
    __syncthreads();
    if (n == 0) {
        float v = crb2[0];
        for (int k = 0; k < 128; ++k) v += h1[k] * crW2[k];
        out[0] = v;
    }
    __syncthreads();

    // ---- action-type head (at)
    a = atb0[n];
    for (int k = 0; k < 144; ++k) a += g[k] * atW0[k * 128 + n];
    h0[n] = fmaxf(a, 0.f);
    __syncthreads();
    a = atb1[n];
    for (int k = 0; k < 128; ++k) a += h0[k] * atW1[k * 128 + n];
    h1[n] = fmaxf(a, 0.f);
    __syncthreads();
    if (n < 6) {
        float v = atb2[n];
        for (int k = 0; k < 128; ++k) v += h1[k] * atW2[k * 6 + n];
        out[1 + n] = v;
    }
}

// ---------------------------------------------------------------------------
// Final mt layer: per-node dot(h(128 f16), W2(128 f32)) + b -> out[7+i]
// ---------------------------------------------------------------------------
__global__ void mt_final_kernel(const h16* __restrict__ h, const float* __restrict__ W2,
                                const float* __restrict__ b2, float* __restrict__ out)
{
    __shared__ float w[128];
    if (threadIdx.x < 128) w[threadIdx.x] = W2[threadIdx.x];
    __syncthreads();
    int i = blockIdx.x * blockDim.x + threadIdx.x;
    if (i >= NNODES) return;
    const h16* p = h + (size_t)i * 128;
    float a = b2[0];
#pragma unroll 8
    for (int k = 0; k < 128; ++k) a += (float)p[k] * w[k];
    out[7 + i] = a;
}

// ---------------------------------------------------------------------------
extern "C" void kernel_launch(void* const* d_in, const int* in_sizes, int n_in,
                              void* d_out, int out_size, void* d_ws, size_t ws_size,
                              hipStream_t stream)
{
    (void)in_sizes; (void)n_in; (void)out_size; (void)ws_size;
    const float* x     = (const float*)d_in[0];
    const float* sWl1  = (const float*)d_in[3];
    const float* sWr1  = (const float*)d_in[4];
    const float* sb1   = (const float*)d_in[5];
    const float* sWl2  = (const float*)d_in[6];
    const float* sWr2  = (const float*)d_in[7];
    const float* sb2   = (const float*)d_in[8];
    const float* sWl3  = (const float*)d_in[9];
    const float* sWr3  = (const float*)d_in[10];
    const float* sb3   = (const float*)d_in[11];
    const float* atW0  = (const float*)d_in[12]; const float* atb0 = (const float*)d_in[13];
    const float* atW1  = (const float*)d_in[14]; const float* atb1 = (const float*)d_in[15];
    const float* atW2  = (const float*)d_in[16]; const float* atb2 = (const float*)d_in[17];
    const float* crW0  = (const float*)d_in[18]; const float* crb0 = (const float*)d_in[19];
    const float* crW1  = (const float*)d_in[20]; const float* crb1 = (const float*)d_in[21];
    const float* crW2  = (const float*)d_in[22]; const float* crb2 = (const float*)d_in[23];
    const float* mtW0  = (const float*)d_in[24]; const float* mtb0 = (const float*)d_in[25];
    const float* mtW1  = (const float*)d_in[26]; const float* mtb1 = (const float*)d_in[27];
    const float* mtW2  = (const float*)d_in[28]; const float* mtb2 = (const float*)d_in[29];
    float* out = (float*)d_out;

    const size_t N = NNODES;
    size_t off = 0;
    char* base = (char*)d_ws;
    auto alloc = [&](size_t bytes) -> char* {
        char* p = base + off;
        off += (bytes + 255) & ~(size_t)255;
        return p;
    };
    h16* bufA1  = (h16*)alloc(N * 64 * 2);    // [agg_x(32) | x16(32)]
    h16* bufH1  = (h16*)alloc(N * 128 * 2);   // h1 -> h3 (GEMM3 out)
    h16* bufH2  = (h16*)alloc(N * 128 * 2);   // h2 -> mt_h1
    h16* bufAgg = (h16*)alloc(N * 128 * 2);   // stencil agg -> mt_h2
    h16* bufPE  = (h16*)alloc(N * 32 * 2);    // [pe(16) | pad(16)]
    h16* sWt1   = (h16*)alloc(128 * 64 * 2);
    h16* sWt2   = (h16*)alloc(128 * 256 * 2);
    h16* sWt3   = (h16*)alloc(128 * 256 * 2);
    h16* mtWt0  = (h16*)alloc(128 * 160 * 2);
    h16* mtWt1  = (h16*)alloc(128 * 128 * 2);
    float* gvec = (float*)alloc(144 * 4);

    // ---- pack weights to f16 W^T (K padded to x32)
    pack_wt_kernel<<<(128 * 64 + 255) / 256, 256, 0, stream>>>(sWl1, 26, 32, sWr1, 26, 32, sWt1);
    pack_wt_kernel<<<(128 * 256 + 255) / 256, 256, 0, stream>>>(sWl2, 128, 128, sWr2, 128, 128, sWt2);
    pack_wt_kernel<<<(128 * 256 + 255) / 256, 256, 0, stream>>>(sWl3, 128, 128, sWr3, 128, 128, sWt3);
    pack_wt_kernel<<<(128 * 160 + 255) / 256, 256, 0, stream>>>(mtW0, 144, 160, nullptr, 0, 0, mtWt0);
    pack_wt_kernel<<<(128 * 128 + 255) / 256, 256, 0, stream>>>(mtW1, 128, 128, nullptr, 0, 0, mtWt1);

    // ---- input conversion + positional encoding
    convert_x_kernel<<<(int)(N * 32 / 256), 256, 0, stream>>>(x, bufA1);
    pe_kernel<<<(int)(N * 32 / 256), 256, 0, stream>>>(bufPE);

    const int gemmBlocks = NNODES / 128;

    // ---- SAGE layer 1 (K = 32 agg + 32 x)
    stencil_kernel<<<(int)(N * 4 / 256), 256, 0, stream>>>(bufA1 + 32, 64, bufA1, 64, 32);
    gemm_wmma_kernel<<<gemmBlocks, 256, 128 * 64 * 2, stream>>>(bufA1, 64, nullptr, 0, sWt1, sb1, bufH1, 128, 1);

    // ---- SAGE layer 2 (K = 128 agg + 128 h)
    stencil_kernel<<<(int)(N * 16 / 256), 256, 0, stream>>>(bufH1, 128, bufAgg, 128, 128);
    gemm_wmma_kernel<<<gemmBlocks, 256, 128 * 256 * 2, stream>>>(bufAgg, 128, bufH1, 128, sWt2, sb2, bufH2, 128, 1);

    // ---- SAGE layer 3
    stencil_kernel<<<(int)(N * 16 / 256), 256, 0, stream>>>(bufH2, 128, bufAgg, 128, 128);
    gemm_wmma_kernel<<<gemmBlocks, 256, 128 * 256 * 2, stream>>>(bufAgg, 128, bufH2, 128, sWt3, sb3, bufH1, 128, 1);

    // ---- mt head layer 0: [h3(128) | pe(32 padded)] -> 128, relu
    gemm_wmma_kernel<<<gemmBlocks, 256, 128 * 160 * 2, stream>>>(bufH1, 128, bufPE, 32, mtWt0, mtb0, bufH2, 128, 1);
    // ---- mt head layer 1: 128 -> 128, relu
    gemm_wmma_kernel<<<gemmBlocks, 256, 128 * 128 * 2, stream>>>(bufH2, 128, nullptr, 0, mtWt1, mtb1, bufAgg, 128, 1);

    // ---- pooled heads
    mean_kernel<<<144, 256, 0, stream>>>(bufH1, bufPE, gvec);
    heads_kernel<<<1, 128, 0, stream>>>(gvec,
                                        atW0, atb0, atW1, atb1, atW2, atb2,
                                        crW0, crb0, crW1, crb1, crW2, crb2, out);

    // ---- final per-node mt logit
    mt_final_kernel<<<(int)(N / 256), 256, 0, stream>>>(bufAgg, mtW2, mtb2, out);
}
